// LocationBasedSocialNetworkGAT_5858335392469
// MI455X (gfx1250) — matrix operand order repaired and checked by reading, not compiled, not run on hardware
//
#include <hip/hip_runtime.h>
#include <float.h>

typedef __attribute__((ext_vector_type(16))) _Float16 v16h;
typedef __attribute__((ext_vector_type(8)))  _Float16 v8h;
typedef __attribute__((ext_vector_type(8)))  float    v8f;

#define HC 64          // H*C = 64 for both layers (2 heads x 32 channels)
#define NEG_SLOPE 0.2f
#define KMAX 128       // max K across layers (layer1: 128, layer2: 64)

// ---------- float <-> order-preserving uint (for native u32 atomic max) ----------
__device__ __forceinline__ unsigned f2o(float f) {
    unsigned u = __float_as_uint(f);
    return (u >> 31) ? ~u : (u | 0x80000000u);
}
__device__ __forceinline__ float o2f(unsigned u) {
    return (u >> 31) ? __uint_as_float(u & 0x7fffffffu) : __uint_as_float(~u);
}

// ---------- init kernels ----------
__global__ void gat_init_md(unsigned* __restrict__ mbuf, float* __restrict__ denom, int nNodes) {
    int t = blockIdx.x * blockDim.x + threadIdx.x;
    if (t >= nNodes * 2) return;
    mbuf[t]  = f2o(-FLT_MAX);
    denom[t] = 0.0f;
}

__global__ void gat_init_out(float* __restrict__ out, const float* __restrict__ bias, int nNodes) {
    int t = blockIdx.x * blockDim.x + threadIdx.x;
    if (t >= nNodes * HC) return;
    out[t] = bias[t & (HC - 1)];
}

// ---------- f32 -> f16 activation convert (ReLU folded for layer 2) ----------
__global__ void gat_cvt_a(const float* __restrict__ in, _Float16* __restrict__ outh,
                          long long n, int applyRelu) {
    long long t = (long long)blockIdx.x * blockDim.x + threadIdx.x;
    if (t >= n) return;
    float v = in[t];
    if (applyRelu) v = v > 0.f ? v : 0.f;
    outh[t] = (_Float16)v;
}

// ---------- W[K,64] f32 -> Wt[64,K] f16 (transpose so B frags are contiguous) ----------
__global__ void gat_cvt_wt(const float* __restrict__ W, _Float16* __restrict__ Wt, int K) {
    int t = blockIdx.x * blockDim.x + threadIdx.x;
    if (t >= HC * K) return;
    int n = t / K, k = t - n * K;
    Wt[t] = (_Float16)W[(size_t)k * HC + n];
}

// ---------- WMMA GEMM: out[nRows,64] = Ah[nRows,K] @ Wt^T, f16 in / f32 acc ----------
// One wave per 16x16 output tile. Wt staged in LDS once per block; A via b128 loads.
__global__ void gat_gemm_wmma(const _Float16* __restrict__ Ah, const _Float16* __restrict__ Wt,
                              float* __restrict__ out, int nRows, int K) {
    __shared__ __align__(16) _Float16 sW[HC * KMAX];   // 16 KB max

    // cooperative LDS fill: Wt is HC*K halfs, 16B chunks
    const int totChunks = (HC * K) / 8;
    for (int i = threadIdx.x; i < totChunks; i += blockDim.x)
        *(v8h*)(sW + i * 8) = *(const v8h*)(Wt + i * 8);
    __syncthreads();

    const int lane   = threadIdx.x & 31;
    const int wave   = threadIdx.x >> 5;
    const int nTiles = HC / 16;                       // 4 column tiles
    const int numTiles = (nRows / 16) * nTiles;
    int tile = blockIdx.x * (blockDim.x >> 5) + wave; // wave-uniform
    if (tile >= numTiles) return;                     // whole wave exits together

    const int mTile  = tile / nTiles;
    const int nTile  = tile % nTiles;
    const int laneLo = lane & 15;
    const int laneHi = lane >> 4;

    const _Float16* arow = Ah + (size_t)(mTile * 16 + laneLo) * K;
    const int       ncol = nTile * 16 + laneLo;
    const _Float16* brow = sW + (size_t)ncol * K;

    v8f acc = {};
    for (int k0 = 0; k0 < K; k0 += 32) {
        if (k0 + 32 < K) __builtin_prefetch(arow + k0 + 32, 0, 3); // global_prefetch_b8

        // A frag (16-bit A 16x32, ISA 7.12.2): lane half laneHi=0 holds K=0..7,16..23;
        // laneHi=1 holds K=8..15,24..31 -> two contiguous 16B global loads.
        const int ka = k0 + (laneHi ? 8 : 0);
        union { v16h v; v8h h[2]; } a;
        a.h[0] = *(const v8h*)(arow + ka);        // K = ka .. ka+7
        a.h[1] = *(const v8h*)(arow + ka + 16);   // K = ka+16 .. ka+23

        // B frag (16-bit B 32x16): same K order as A: lanes 0-15 hold K=0..7,16..23, lanes 16-31 K=8..15,24..31,
        // column = laneLo -> two 16B LDS loads from transposed weights.
        const int kb = k0 + (laneHi ? 8 : 0);
        union { v16h v; v8h h[2]; } b;
        b.h[0] = *(const v8h*)(brow + kb);
        b.h[1] = *(const v8h*)(brow + kb + 16);

        acc = __builtin_amdgcn_wmma_f32_16x16x32_f16(
            /*neg_a=*/false, a.v, /*neg_b=*/false, b.v,
            /*c_mod=*/(short)0, acc, /*reuse_a=*/false, /*reuse_b=*/false);
    }

    // D layout: VGPR r -> row mTile*16 + r + 8*laneHi, col = ncol
#pragma unroll
    for (int r = 0; r < 8; ++r) {
        int m = mTile * 16 + r + 8 * laneHi;
        out[(size_t)m * HC + ncol] = acc[r];
    }
}

// ---------- per-node attention scores: a_s[n,h], a_d[n,h] ----------
__global__ void gat_att(const float* __restrict__ h, const float* __restrict__ att_src,
                        const float* __restrict__ att_dst,
                        float* __restrict__ as_, float* __restrict__ ad_, int nNodes) {
    int t = blockIdx.x * blockDim.x + threadIdx.x;
    if (t >= nNodes * 2) return;
    int n = t >> 1, hd = t & 1;
    const float* hp = h + (size_t)n * HC + hd * 32;
    float s = 0.f, d = 0.f;
#pragma unroll
    for (int c = 0; c < 32; ++c) {
        float v = hp[c];
        s += v * att_src[hd * 32 + c];
        d += v * att_dst[hd * 32 + c];
    }
    as_[t] = s;
    ad_[t] = d;
}

// ---------- edge pass 1: e = leaky_relu(a_s[src]+a_d[dst]); segment max ----------
__global__ void gat_edge_score(const int* __restrict__ src, const int* __restrict__ dst,
                               const float* __restrict__ as_, const float* __restrict__ ad_,
                               float* __restrict__ ebuf, unsigned* __restrict__ mbuf, int nEdges) {
    int t = blockIdx.x * blockDim.x + threadIdx.x;
    if (t >= nEdges * 2) return;
    int e = t >> 1, hd = t & 1;
    int s = src[e], d = dst[e];
    float v = as_[s * 2 + hd] + ad_[d * 2 + hd];
    v = v > 0.f ? v : NEG_SLOPE * v;
    ebuf[t] = v;
    atomicMax(&mbuf[d * 2 + hd], f2o(v));   // native global_atomic_max_u32
}

// ---------- edge pass 2: ex = exp(e - m[dst]); segment sum of ex ----------
__global__ void gat_edge_exp(const int* __restrict__ dst, const unsigned* __restrict__ mbuf,
                             float* __restrict__ ebuf, float* __restrict__ denom, int nEdges) {
    int t = blockIdx.x * blockDim.x + threadIdx.x;
    if (t >= nEdges * 2) return;
    int e = t >> 1, hd = t & 1;
    int d = dst[e];
    float ex = __expf(ebuf[t] - o2f(mbuf[d * 2 + hd]));
    ebuf[t] = ex;
    atomicAdd(&denom[d * 2 + hd], ex);      // global_atomic_add_f32
}

// ---------- edge pass 3: out[dst] += h[src] * (ex/denom[dst]) ----------
// 4 threads per edge, 16 features each (float4 gathers; L2-resident on MI455X).
__global__ void gat_aggregate(const int* __restrict__ src, const int* __restrict__ dst,
                              const float* __restrict__ h, const float* __restrict__ ebuf,
                              const float* __restrict__ denom, float* __restrict__ out, int nEdges) {
    int t = blockIdx.x * blockDim.x + threadIdx.x;
    if (t >= nEdges * 4) return;
    int e = t >> 2, q = t & 3, hd = q >> 1;
    int s = src[e], d = dst[e];
    float alpha = ebuf[e * 2 + hd] / denom[d * 2 + hd];
    const float4* hs = (const float4*)(h + (size_t)s * HC + q * 16);
    float*        od = out + (size_t)d * HC + q * 16;
#pragma unroll
    for (int i = 0; i < 4; ++i) {
        float4 v = hs[i];
        atomicAdd(od + 4 * i + 0, v.x * alpha);
        atomicAdd(od + 4 * i + 1, v.y * alpha);
        atomicAdd(od + 4 * i + 2, v.z * alpha);
        atomicAdd(od + 4 * i + 3, v.w * alpha);
    }
}

// =======================================================================
static inline int cdiv(long long a, int b) { return (int)((a + b - 1) / b); }

extern "C" void kernel_launch(void* const* d_in, const int* in_sizes, int n_in,
                              void* d_out, int out_size, void* d_ws, size_t ws_size,
                              hipStream_t stream) {
    const int F_IN = 128;
    const int N = in_sizes[0] / F_IN;   // 50000
    const int E = in_sizes[1] / 2;      // 1,600,000

    const float* x    = (const float*)d_in[0];
    const int*   ei   = (const int*)  d_in[1];
    const int*   srcI = ei;
    const int*   dstI = ei + E;
    const float* W1   = (const float*)d_in[2];
    const float* as1  = (const float*)d_in[3];
    const float* ad1  = (const float*)d_in[4];
    const float* b1   = (const float*)d_in[5];
    const float* W2   = (const float*)d_in[6];
    const float* as2  = (const float*)d_in[7];
    const float* ad2  = (const float*)d_in[8];
    const float* b2   = (const float*)d_in[9];
    float*       outF = (float*)d_out;

    // workspace carve-up (all generously aligned by construction)
    char* w = (char*)d_ws;
    float*     hbuf  = (float*)w;     w += (size_t)N * HC * sizeof(float);
    float*     out1  = (float*)w;     w += (size_t)N * HC * sizeof(float);
    float*     ebuf  = (float*)w;     w += (size_t)E * 2  * sizeof(float);
    _Float16*  Ah    = (_Float16*)w;  w += (size_t)N * KMAX * sizeof(_Float16);
    _Float16*  Wt    = (_Float16*)w;  w += (size_t)HC * KMAX * sizeof(_Float16);
    float*     asrc  = (float*)w;     w += (size_t)N * 2  * sizeof(float);
    float*     adst  = (float*)w;     w += (size_t)N * 2  * sizeof(float);
    unsigned*  mbuf  = (unsigned*)w;  w += (size_t)N * 2  * sizeof(unsigned);
    float*     denom = (float*)w;     w += (size_t)N * 2  * sizeof(float);
    (void)ws_size; (void)n_in; (void)out_size;

    const int B = 256;
    const int wavesPerBlock = B / 32;
    const int numTiles = (N / 16) * (HC / 16);
    const int gemmGrid = cdiv(numTiles, wavesPerBlock);

    // ===================== Layer 1 =====================
    gat_init_md <<<cdiv((long long)N * 2,  B), B, 0, stream>>>(mbuf, denom, N);
    gat_init_out<<<cdiv((long long)N * HC, B), B, 0, stream>>>(out1, b1, N);
    gat_cvt_a   <<<cdiv((long long)N * F_IN, B), B, 0, stream>>>(x, Ah, (long long)N * F_IN, 0);
    gat_cvt_wt  <<<cdiv((long long)HC * F_IN, B), B, 0, stream>>>(W1, Wt, F_IN);
    gat_gemm_wmma<<<gemmGrid, B, 0, stream>>>(Ah, Wt, hbuf, N, F_IN);
    gat_att     <<<cdiv((long long)N * 2,  B), B, 0, stream>>>(hbuf, as1, ad1, asrc, adst, N);
    gat_edge_score<<<cdiv((long long)E * 2, B), B, 0, stream>>>(srcI, dstI, asrc, adst, ebuf, mbuf, E);
    gat_edge_exp  <<<cdiv((long long)E * 2, B), B, 0, stream>>>(dstI, mbuf, ebuf, denom, E);
    gat_aggregate <<<cdiv((long long)E * 4, B), B, 0, stream>>>(srcI, dstI, hbuf, ebuf, denom, out1, E);

    // ===================== Layer 2 (ReLU folded into f16 convert) ====
    gat_init_md <<<cdiv((long long)N * 2,  B), B, 0, stream>>>(mbuf, denom, N);
    gat_init_out<<<cdiv((long long)N * HC, B), B, 0, stream>>>(outF, b2, N);
    gat_cvt_a   <<<cdiv((long long)N * HC, B), B, 0, stream>>>(out1, Ah, (long long)N * HC, 1);
    gat_cvt_wt  <<<cdiv((long long)HC * HC, B), B, 0, stream>>>(W2, Wt, HC);
    gat_gemm_wmma<<<gemmGrid, B, 0, stream>>>(Ah, Wt, hbuf, N, HC);
    gat_att     <<<cdiv((long long)N * 2,  B), B, 0, stream>>>(hbuf, as2, ad2, asrc, adst, N);
    gat_edge_score<<<cdiv((long long)E * 2, B), B, 0, stream>>>(srcI, dstI, asrc, adst, ebuf, mbuf, E);
    gat_edge_exp  <<<cdiv((long long)E * 2, B), B, 0, stream>>>(dstI, mbuf, ebuf, denom, E);
    gat_aggregate <<<cdiv((long long)E * 4, B), B, 0, stream>>>(srcI, dstI, hbuf, ebuf, denom, outF, E);
}
